// DoubleStreamBlock_7327214207286
// MI455X (gfx1250) — compile-verified, matrix operand order checked by reading, not run on hardware
//
#include <hip/hip_runtime.h>

// ---------- types ----------
typedef __bf16 bf16;
typedef __attribute__((ext_vector_type(16))) __bf16 v16bf;
typedef __attribute__((ext_vector_type(8)))  float  v8f;

#define NB   2
#define HIDD 3072
#define NHEAD 24
#define HDIM 128
#define MLPD 12288
#define TXTL 5
#define IMGS 2048
#define LSEQ 2053
#define LPAD 2112
#define PEL  4101   // TXT_L + IMG_L

__device__ __forceinline__ v16bf ld_frag(const bf16* p, int hl) {
  union { uint4 u[2]; v16bf v; } t;
  t.u[0] = *(const uint4*)(p + hl * 8);
  t.u[1] = *(const uint4*)(p + 16 + hl * 8);
  return t.v;
}

__device__ __forceinline__ v8f wmma_bf16(v16bf a, v16bf b, v8f c) {
  return __builtin_amdgcn_wmma_f32_16x16x32_bf16(false, a, false, b, (short)0, c, false, false);
}

__device__ __forceinline__ float gelu_t(float x) {
  float x3 = x * x * x;
  return 0.5f * x * (1.f + tanhf(0.7978845608028654f * (x + 0.044715f * x3)));
}

// ---------- tiny elementwise kernels ----------
__global__ void silu_k(const float* __restrict__ v, float* __restrict__ o, int n) {
  int i = blockIdx.x * 256 + threadIdx.x;
  if (i < n) { float x = v[i]; o[i] = x / (1.f + __expf(-x)); }
}

// out[row,c] = base + g[c]*val[row,c] ; mode 0 = img-slice extraction base, 1 = plain base
__global__ void residual_k(const float* __restrict__ base, const float* __restrict__ val,
                           const float* __restrict__ modv, int gCh,
                           float* __restrict__ out, int rowsPerB, int mode, int total) {
  int idx = blockIdx.x * 256 + threadIdx.x;
  if (idx >= total) return;
  int c = idx % HIDD;
  int row = idx / HIDD;
  int b = row / rowsPerB, l = row % rowsPerB;
  float bs;
  if (mode == 0) {
    int h2 = l >> 5, w2 = l & 31;
    bs = base[((size_t)((b * 64 + h2) * 64 + w2)) * HIDD + c];
  } else {
    bs = base[(size_t)row * HIDD + c];
  }
  float g = modv[(size_t)b * (6 * HIDD) + (size_t)gCh * HIDD + c];
  out[(size_t)row * HIDD + c] = bs + g * val[(size_t)row * HIDD + c];
}

// ---------- LayerNorm + modulation: out = (1+scale)*ln(x) + shift ----------
__global__ void __launch_bounds__(256)
ln_mod_k(const float* __restrict__ x, const float* __restrict__ modv,
         float* __restrict__ out, int rowsPerB, int mode, int shCh, int scCh) {
  __shared__ float s1[256], s2[256];
  const int row = blockIdx.x;
  const int b = row / rowsPerB;
  const int l = row % rowsPerB;
  const int tid = threadIdx.x;
  const float* src;
  if (mode == 1) {
    const int h2 = l >> 5, w2 = l & 31;
    src = x + ((size_t)((b * 64 + h2) * 64 + w2)) * HIDD;
  } else {
    src = x + (size_t)row * HIDD;
  }
  float xv[12], sa = 0.f, sb = 0.f;
#pragma unroll
  for (int i = 0; i < 12; ++i) {
    float v = src[tid + i * 256];
    xv[i] = v; sa += v; sb += v * v;
  }
  s1[tid] = sa; s2[tid] = sb; __syncthreads();
  for (int s = 128; s > 0; s >>= 1) {
    if (tid < s) { s1[tid] += s1[tid + s]; s2[tid] += s2[tid + s]; }
    __syncthreads();
  }
  const float mean = s1[0] * (1.f / (float)HIDD);
  const float var  = s2[0] * (1.f / (float)HIDD) - mean * mean;
  const float r = rsqrtf(var + 1e-6f);
  const float* mb = modv + (size_t)b * (6 * HIDD);
#pragma unroll
  for (int i = 0; i < 12; ++i) {
    int c = tid + i * 256;
    float sh = mb[(size_t)shCh * HIDD + c];
    float sc = mb[(size_t)scCh * HIDD + c];
    out[(size_t)row * HIDD + c] = (1.f + sc) * ((xv[i] - mean) * r) + sh;
  }
}

// ---------- generic bf16-WMMA GEMM: out[M,N] = A[M,K] @ W[K,N] + bias, epi 1 => gelu ----------
__global__ void __launch_bounds__(256)
gemm_bf16_k(const float* __restrict__ A, const float* __restrict__ W,
            const float* __restrict__ bias, float* __restrict__ out,
            int M, int N, int K, int epi) {
  __shared__ __align__(16) bf16 As[128][32];   // [m][k]
  __shared__ __align__(16) bf16 Bs[128][32];   // [n][k] (transposed on store)
  const int tid  = threadIdx.x;
  const int lane = tid & 31;
  const int wv   = tid >> 5;
  const int wm   = wv & 3;     // 4 waves along M (32 rows each)
  const int wn   = wv >> 2;    // 2 waves along N (64 cols each)
  const int hl   = lane >> 4;
  const int lm   = lane & 15;
  const int m0 = blockIdx.y * 128;
  const int n0 = blockIdx.x * 128;

  v8f acc[2][4];
#pragma unroll
  for (int i = 0; i < 2; ++i)
#pragma unroll
    for (int j = 0; j < 4; ++j)
      acc[i][j] = (v8f){0.f, 0.f, 0.f, 0.f, 0.f, 0.f, 0.f, 0.f};

  const int aRow = tid >> 1;
  const int aSeg = (tid & 1) * 16;
  const int bK   = tid >> 3;
  const int bSeg = (tid & 7) * 16;
  // Clamp the A row: rows >= M only feed output rows >= M, which are never
  // stored, so reading a valid row instead is harmless and branch-free.
  const int gmc  = (m0 + aRow < M) ? (m0 + aRow) : (M - 1);
  const float* ap = A + (size_t)gmc * K + aSeg;
  const float* wp = W + (size_t)bK * N + n0 + bSeg;

  for (int k0 = 0; k0 < K; k0 += 32) {
    // issue all 8 b128 loads first (keeps them in flight together)
    float4 fa[4], fw[4];
#pragma unroll
    for (int c = 0; c < 4; ++c) fa[c] = *(const float4*)(ap + k0 + c * 4);
#pragma unroll
    for (int c = 0; c < 4; ++c) fw[c] = *(const float4*)(wp + (size_t)k0 * N + c * 4);
    // fp32 -> bf16 -> LDS (A rows contiguous: compiler merges to ds_store_b64)
#pragma unroll
    for (int c = 0; c < 4; ++c) {
      As[aRow][aSeg + c * 4 + 0] = (bf16)fa[c].x;
      As[aRow][aSeg + c * 4 + 1] = (bf16)fa[c].y;
      As[aRow][aSeg + c * 4 + 2] = (bf16)fa[c].z;
      As[aRow][aSeg + c * 4 + 3] = (bf16)fa[c].w;
      Bs[bSeg + c * 4 + 0][bK] = (bf16)fw[c].x;
      Bs[bSeg + c * 4 + 1][bK] = (bf16)fw[c].y;
      Bs[bSeg + c * 4 + 2][bK] = (bf16)fw[c].z;
      Bs[bSeg + c * 4 + 3][bK] = (bf16)fw[c].w;
    }
    __syncthreads();
    if (k0 + 32 < K) {   // prefetch next tile (global_prefetch_b8)
      __builtin_prefetch(ap + k0 + 32, 0, 1);
      __builtin_prefetch(wp + (size_t)(k0 + 32) * N, 0, 1);
    }
    v16bf af[2], bfr[4];
#pragma unroll
    for (int i = 0; i < 2; ++i) af[i]  = ld_frag(&As[wm * 32 + i * 16 + lm][0], hl);
#pragma unroll
    for (int j = 0; j < 4; ++j) bfr[j] = ld_frag(&Bs[wn * 64 + j * 16 + lm][0], hl);
#pragma unroll
    for (int i = 0; i < 2; ++i)
#pragma unroll
      for (int j = 0; j < 4; ++j)
        acc[i][j] = wmma_bf16(af[i], bfr[j], acc[i][j]);
    __syncthreads();
  }
  // epilogue: C layout -> lane 0-15: M=r, N=lane ; lane 16-31: M=8+r
#pragma unroll
  for (int i = 0; i < 2; ++i) {
#pragma unroll
    for (int j = 0; j < 4; ++j) {
      const int col = n0 + wn * 64 + j * 16 + lm;
      const float bv = bias[col];
#pragma unroll
      for (int e = 0; e < 8; ++e) {
        const int row = m0 + wm * 32 + i * 16 + hl * 8 + e;
        if (row < M) {
          float v = acc[i][j][e] + bv;
          if (epi == 1) v = gelu_t(v);
          out[(size_t)row * N + col] = v;
        }
      }
    }
  }
}

// ---------- qkv postprocess: RMS norm + RoPE ----------
// q,k packed (B,H,LPAD,HD); v packed TRANSPOSED (B,H,HD,LPAD) so attention can
// load P@V B-fragments straight from global memory (contiguous along keys).
__global__ void __launch_bounds__(128)
qkv_post_k(const float* __restrict__ qkv, const float* __restrict__ qs,
           const float* __restrict__ ks, const float* __restrict__ pe,
           bf16* __restrict__ qo, bf16* __restrict__ ko, bf16* __restrict__ vt,
           int Ls, int lOff) {
  __shared__ float sq[128], sk[128], r1[128], r2[128];
  const int gid = blockIdx.x;
  const int h = gid % NHEAD;
  const int t = gid / NHEAD;
  const int l = t % Ls;
  const int b = t / Ls;
  const int d = threadIdx.x;
  const float* base = qkv + (size_t)(b * Ls + l) * (3 * HIDD);
  float q = base[h * HDIM + d];
  float k = base[HIDD + h * HDIM + d];
  float v = base[2 * HIDD + h * HDIM + d];
  r1[d] = q * q; r2[d] = k * k; __syncthreads();
  for (int s = 64; s > 0; s >>= 1) {
    if (d < s) { r1[d] += r1[d + s]; r2[d] += r2[d + s]; }
    __syncthreads();
  }
  const float qr = rsqrtf(r1[0] * (1.f / (float)HDIM) + 1e-6f);
  const float kr = rsqrtf(r2[0] * (1.f / (float)HDIM) + 1e-6f);
  sq[d] = q * qr * qs[d];
  sk[d] = k * kr * ks[d];
  __syncthreads();
  const int pidx = lOff ? (TXTL + (l >> 5) * 64 + (l & 31)) : l;
  const int d0 = d >> 1, sbit = d & 1;
  const float* pp = pe + ((((size_t)b * PEL + pidx) * 64 + d0) * 2 + sbit) * 2;
  const float qv = pp[0] * sq[2 * d0] + pp[1] * sq[2 * d0 + 1];
  const float kv = pp[0] * sk[2 * d0] + pp[1] * sk[2 * d0 + 1];
  const int pos = lOff + l;
  const int bh = b * NHEAD + h;
  const size_t o = ((size_t)bh * LPAD + pos) * HDIM + d;
  qo[o] = (bf16)qv; ko[o] = (bf16)kv;
  vt[((size_t)bh * HDIM + d) * LPAD + pos] = (bf16)v;
}

__global__ void pad_qkv_k(bf16* __restrict__ qo, bf16* __restrict__ ko, bf16* __restrict__ vt) {
  const int idx = blockIdx.x;
  const int r = idx % (LPAD - LSEQ);
  const int bh = idx / (LPAD - LSEQ);
  const int d = threadIdx.x;
  const size_t o = ((size_t)bh * LPAD + LSEQ + r) * HDIM + d;
  qo[o] = (bf16)0.f; ko[o] = (bf16)0.f;
  vt[((size_t)bh * HDIM + d) * LPAD + LSEQ + r] = (bf16)0.f;
}

// ---------- flash attention: grid (33 q-tiles, B*H), 4 waves * 16 q rows ----------
// No block barriers: K and V(transposed) fragments come straight from global;
// only the per-wave P transpose goes through LDS.
__global__ void __launch_bounds__(128)
attention_k(const bf16* __restrict__ Q, const bf16* __restrict__ Kb,
            const bf16* __restrict__ Vt, float* __restrict__ outImg,
            float* __restrict__ outTxt) {
  __shared__ __align__(16) bf16 Ps[4][16][64];   // per-wave P tile
  const int bh = blockIdx.y;
  const int b = bh / NHEAD, h = bh % NHEAD;
  const int qt = blockIdx.x;
  const int tid = threadIdx.x, lane = tid & 31, wv = tid >> 5;
  const int hl = lane >> 4, lm = lane & 15;

  const bf16* qp = Q + ((size_t)bh * LPAD + qt * 64 + wv * 16 + lm) * HDIM;
  v16bf qf[4];
#pragma unroll
  for (int kd = 0; kd < 4; ++kd) qf[kd] = ld_frag(qp + kd * 32, hl);

  v8f o[8];
  float mrun[8], lrun[8];
#pragma unroll
  for (int e = 0; e < 8; ++e) { mrun[e] = -1e30f; lrun[e] = 0.f; }
#pragma unroll
  for (int jd = 0; jd < 8; ++jd) o[jd] = (v8f){0.f, 0.f, 0.f, 0.f, 0.f, 0.f, 0.f, 0.f};

  for (int kt = 0; kt < LPAD / 64; ++kt) {
    // S = Q @ K^T  (K fragments straight from global: contiguous in d)
    v8f s[4];
#pragma unroll
    for (int j = 0; j < 4; ++j) s[j] = (v8f){0.f, 0.f, 0.f, 0.f, 0.f, 0.f, 0.f, 0.f};
#pragma unroll
    for (int kd = 0; kd < 4; ++kd) {
#pragma unroll
      for (int j = 0; j < 4; ++j) {
        const bf16* kp = Kb + ((size_t)bh * LPAD + kt * 64 + j * 16 + lm) * HDIM + kd * 32;
        v16bf kf = ld_frag(kp, hl);
        s[j] = wmma_bf16(qf[kd], kf, s[j]);
      }
    }
    const float scl = 0.08838834764831845f;  // 1/sqrt(128)
#pragma unroll
    for (int j = 0; j < 4; ++j) {
      const int key = kt * 64 + j * 16 + lm;
      const float msk = (key < LSEQ) ? 0.f : -1e30f;
#pragma unroll
      for (int e = 0; e < 8; ++e) s[j][e] = s[j][e] * scl + msk;
    }
    // online softmax (rows live in 16-lane halves, matching C-matrix layout)
#pragma unroll
    for (int e = 0; e < 8; ++e) {
      float mt = fmaxf(fmaxf(s[0][e], s[1][e]), fmaxf(s[2][e], s[3][e]));
#pragma unroll
      for (int m = 1; m < 16; m <<= 1) mt = fmaxf(mt, __shfl_xor(mt, m, 32));
      const float mn = fmaxf(mrun[e], mt);
      const float alpha = __expf(mrun[e] - mn);
      mrun[e] = mn;
      float rs = 0.f;
#pragma unroll
      for (int j = 0; j < 4; ++j) { float p = __expf(s[j][e] - mn); s[j][e] = p; rs += p; }
#pragma unroll
      for (int m = 1; m < 16; m <<= 1) rs += __shfl_xor(rs, m, 32);
      lrun[e] = lrun[e] * alpha + rs;
#pragma unroll
      for (int jd = 0; jd < 8; ++jd) o[jd][e] *= alpha;
    }
    // P -> per-wave LDS (transpose C-layout to A-frag layout; wave-private,
    // LDS ops are in-order within a wave so no barrier is needed)
#pragma unroll
    for (int j = 0; j < 4; ++j)
#pragma unroll
      for (int e = 0; e < 8; ++e)
        Ps[wv][hl * 8 + e][j * 16 + lm] = (bf16)s[j][e];
    // O += P @ V  (V^T fragments straight from global: contiguous in keys)
#pragma unroll
    for (int kk = 0; kk < 2; ++kk) {
      v16bf pf = ld_frag(&Ps[wv][lm][kk * 32], hl);
#pragma unroll
      for (int jd = 0; jd < 8; ++jd) {
        const bf16* vp = Vt + ((size_t)bh * HDIM + jd * 16 + lm) * LPAD + kt * 64 + kk * 32;
        o[jd] = wmma_bf16(pf, ld_frag(vp, hl), o[jd]);
      }
    }
  }
  // write O, split txt/img rows, (b,l,h*128+d) layout
#pragma unroll
  for (int e = 0; e < 8; ++e) {
    const int seq = qt * 64 + wv * 16 + hl * 8 + e;
    if (seq >= LSEQ) continue;
    const float inv = 1.f / lrun[e];
#pragma unroll
    for (int jd = 0; jd < 8; ++jd) {
      const int col = h * HDIM + jd * 16 + lm;
      const float vvv = o[jd][e] * inv;
      if (seq < TXTL) outTxt[((size_t)b * TXTL + seq) * HIDD + col] = vvv;
      else            outImg[((size_t)b * IMGS + (seq - TXTL)) * HIDD + col] = vvv;
    }
  }
}

// ---------- host-side orchestration ----------
extern "C" void kernel_launch(void* const* d_in, const int* in_sizes, int n_in,
                              void* d_out, int out_size, void* d_ws, size_t ws_size,
                              hipStream_t stream) {
  (void)in_sizes; (void)n_in; (void)out_size; (void)ws_size;
  const float* img = (const float*)d_in[0];
  const float* txt = (const float*)d_in[1];
  const float* vec = (const float*)d_in[2];
  const float* pe  = (const float*)d_in[3];
  const float* img_mod_w  = (const float*)d_in[4];
  const float* img_mod_b  = (const float*)d_in[5];
  const float* img_qkv_w  = (const float*)d_in[6];
  const float* img_qkv_b  = (const float*)d_in[7];
  const float* img_proj_w = (const float*)d_in[8];
  const float* img_proj_b = (const float*)d_in[9];
  const float* img_mlp_w1 = (const float*)d_in[10];
  const float* img_mlp_b1 = (const float*)d_in[11];
  const float* img_mlp_w2 = (const float*)d_in[12];
  const float* img_mlp_b2 = (const float*)d_in[13];
  const float* img_q_scale = (const float*)d_in[14];
  const float* img_k_scale = (const float*)d_in[15];
  const float* txt_mod_w  = (const float*)d_in[16];
  const float* txt_mod_b  = (const float*)d_in[17];
  const float* txt_qkv_w  = (const float*)d_in[18];
  const float* txt_qkv_b  = (const float*)d_in[19];
  const float* txt_proj_w = (const float*)d_in[20];
  const float* txt_proj_b = (const float*)d_in[21];
  const float* txt_mlp_w1 = (const float*)d_in[22];
  const float* txt_mlp_b1 = (const float*)d_in[23];
  const float* txt_mlp_w2 = (const float*)d_in[24];
  const float* txt_mlp_b2 = (const float*)d_in[25];
  const float* txt_q_scale = (const float*)d_in[26];
  const float* txt_k_scale = (const float*)d_in[27];

  float* outImg = (float*)d_out;
  float* outTxt = outImg + (size_t)NB * IMGS * HIDD;

  size_t off = 0;
  auto alloc = [&](size_t bytes) -> void* {
    void* p = (char*)d_ws + off;
    off += (bytes + 255) & ~(size_t)255;
    return p;
  };
  const int MI = NB * IMGS;  // 4096 img rows
  const int MT = NB * TXTL;  // 10 txt rows
  float* sv        = (float*)alloc((size_t)NB * HIDD * 4);
  float* modv_img  = (float*)alloc((size_t)NB * 6 * HIDD * 4);
  float* modv_txt  = (float*)alloc((size_t)NB * 6 * HIDD * 4);
  float* xm_img    = (float*)alloc((size_t)MI * HIDD * 4);
  float* xm_txt    = (float*)alloc((size_t)MT * HIDD * 4);
  float* qkv_img   = (float*)alloc((size_t)MI * 3 * HIDD * 4);
  float* qkv_txt   = (float*)alloc((size_t)MT * 3 * HIDD * 4);
  bf16*  qb        = (bf16*)alloc((size_t)NB * NHEAD * LPAD * HDIM * 2);
  bf16*  kb        = (bf16*)alloc((size_t)NB * NHEAD * LPAD * HDIM * 2);
  bf16*  vtb       = (bf16*)alloc((size_t)NB * NHEAD * LPAD * HDIM * 2);
  float* attn_img  = (float*)alloc((size_t)MI * HIDD * 4);
  float* attn_txt  = (float*)alloc((size_t)MT * HIDD * 4);
  float* h_img     = (float*)alloc((size_t)MI * MLPD * 4);
  float* h_txt     = (float*)alloc((size_t)MT * MLPD * 4);
  float* t1_img    = (float*)alloc((size_t)MI * HIDD * 4);
  float* t1_txt    = (float*)alloc((size_t)MT * HIDD * 4);

  // 1. silu(vec)
  silu_k<<<(NB * HIDD + 255) / 256, 256, 0, stream>>>(vec, sv, NB * HIDD);
  // 2. modulation vectors: silu(vec) @ mod_w + mod_b
  gemm_bf16_k<<<dim3(6 * HIDD / 128, 1), 256, 0, stream>>>(sv, img_mod_w, img_mod_b, modv_img, NB, 6 * HIDD, HIDD, 0);
  gemm_bf16_k<<<dim3(6 * HIDD / 128, 1), 256, 0, stream>>>(sv, txt_mod_w, txt_mod_b, modv_txt, NB, 6 * HIDD, HIDD, 0);
  // 3. LN + mod stage 1
  ln_mod_k<<<MI, 256, 0, stream>>>(img, modv_img, xm_img, IMGS, 1, 0, 1);
  ln_mod_k<<<MT, 256, 0, stream>>>(txt, modv_txt, xm_txt, TXTL, 0, 0, 1);
  // 4. qkv GEMMs
  gemm_bf16_k<<<dim3(3 * HIDD / 128, MI / 128), 256, 0, stream>>>(xm_img, img_qkv_w, img_qkv_b, qkv_img, MI, 3 * HIDD, HIDD, 0);
  gemm_bf16_k<<<dim3(3 * HIDD / 128, 1), 256, 0, stream>>>(xm_txt, txt_qkv_w, txt_qkv_b, qkv_txt, MT, 3 * HIDD, HIDD, 0);
  // 5. RMS + RoPE + pack bf16 (V transposed)
  qkv_post_k<<<MI * NHEAD, 128, 0, stream>>>(qkv_img, img_q_scale, img_k_scale, pe, qb, kb, vtb, IMGS, TXTL);
  qkv_post_k<<<MT * NHEAD, 128, 0, stream>>>(qkv_txt, txt_q_scale, txt_k_scale, pe, qb, kb, vtb, TXTL, 0);
  pad_qkv_k<<<NB * NHEAD * (LPAD - LSEQ), 128, 0, stream>>>(qb, kb, vtb);
  // 6. flash attention
  attention_k<<<dim3(LPAD / 64, NB * NHEAD), 128, 0, stream>>>(qb, kb, vtb, attn_img, attn_txt);
  // 7. proj GEMMs
  gemm_bf16_k<<<dim3(HIDD / 128, MI / 128), 256, 0, stream>>>(attn_img, img_proj_w, img_proj_b, t1_img, MI, HIDD, HIDD, 0);
  gemm_bf16_k<<<dim3(HIDD / 128, 1), 256, 0, stream>>>(attn_txt, txt_proj_w, txt_proj_b, t1_txt, MT, HIDD, HIDD, 0);
  // 8. residual 1 (gate chunk 2) -> d_out is the residual stream
  residual_k<<<(MI * HIDD + 255) / 256, 256, 0, stream>>>(img, t1_img, modv_img, 2, outImg, IMGS, 0, MI * HIDD);
  residual_k<<<(MT * HIDD + 255) / 256, 256, 0, stream>>>(txt, t1_txt, modv_txt, 2, outTxt, TXTL, 1, MT * HIDD);
  // 9. LN + mod stage 2 (chunks 3,4)
  ln_mod_k<<<MI, 256, 0, stream>>>(outImg, modv_img, xm_img, IMGS, 0, 3, 4);
  ln_mod_k<<<MT, 256, 0, stream>>>(outTxt, modv_txt, xm_txt, TXTL, 0, 3, 4);
  // 10. mlp1 (+gelu epilogue)
  gemm_bf16_k<<<dim3(MLPD / 128, MI / 128), 256, 0, stream>>>(xm_img, img_mlp_w1, img_mlp_b1, h_img, MI, MLPD, HIDD, 1);
  gemm_bf16_k<<<dim3(MLPD / 128, 1), 256, 0, stream>>>(xm_txt, txt_mlp_w1, txt_mlp_b1, h_txt, MT, MLPD, HIDD, 1);
  // 11. mlp2
  gemm_bf16_k<<<dim3(HIDD / 128, MI / 128), 256, 0, stream>>>(h_img, img_mlp_w2, img_mlp_b2, t1_img, MI, HIDD, MLPD, 0);
  gemm_bf16_k<<<dim3(HIDD / 128, 1), 256, 0, stream>>>(h_txt, txt_mlp_w2, txt_mlp_b2, t1_txt, MT, HIDD, MLPD, 0);
  // 12. residual 2 (gate chunk 5), in-place on d_out
  residual_k<<<(MI * HIDD + 255) / 256, 256, 0, stream>>>(outImg, t1_img, modv_img, 5, outImg, IMGS, 1, MI * HIDD);
  residual_k<<<(MT * HIDD + 255) / 256, 256, 0, stream>>>(outTxt, t1_txt, modv_txt, 5, outTxt, TXTL, 1, MT * HIDD);
}
